// SnapshotGAT_9345848836491
// MI455X (gfx1250) — matrix operand order repaired and checked by reading, not compiled, not run on hardware
//
#include <hip/hip_runtime.h>
#include <hip/hip_bf16.h>

typedef __attribute__((ext_vector_type(16))) _Float16 v16h;
typedef __attribute__((ext_vector_type(8)))  _Float16 v8h;
typedef __attribute__((ext_vector_type(8)))  float    v8f;

#define GAT_N     4096
#define GAT_ALPHA 0.2f

// ---- WMMA fragment helpers (gfx1250 wave32 layouts, CDNA5 ISA) ----
// A operand (16x32 f16): lane L holds row M=L&15; element i maps to
// K = (i<8 ? i : i+8) + 8*(L>=16)  -> two contiguous 8-half (16B) runs.
__device__ __forceinline__ int a_kidx(int i, int hi) {
    return ((i < 8) ? i : (i + 8)) + (hi << 3);
}
// B operand (32x16 f16): lane L holds col N=L&15; element i maps to K = (i<8 ? i : i+8) + 8*(L>=16).
// We pre-swizzle B so each lane's 16 halfs are contiguous (two b128 loads).

__device__ __forceinline__ v16h load16h(const _Float16* p) {
    v8h lo = *(const v8h*)p;
    v8h hi = *(const v8h*)(p + 8);
    return __builtin_shufflevector(lo, hi, 0,1,2,3,4,5,6,7,8,9,10,11,12,13,14,15);
}
__device__ __forceinline__ v16h load16h2(const _Float16* p0, const _Float16* p1) {
    v8h lo = *(const v8h*)p0;
    v8h hi = *(const v8h*)p1;
    return __builtin_shufflevector(lo, hi, 0,1,2,3,4,5,6,7,8,9,10,11,12,13,14,15);
}
#define WMMA_F16(a, b, c) __builtin_amdgcn_wmma_f32_16x16x32_f16(false, (a), false, (b), (short)0, (c), false, false)

// ---- Kernel 0: bit-pack adjacency (64 MB int32 -> 2 MB bitmask), wave32 ballot ----
__global__ void gat_pack_adj(const int* __restrict__ adj, unsigned* __restrict__ bits) {
    int gid   = blockIdx.x * blockDim.x + threadIdx.x;
    int lane  = gid & 31;
    int wave  = gid >> 5;            // 4096*128 waves
    int row   = wave >> 7;
    int c     = wave & 127;
    int v     = adj[(size_t)row * GAT_N + c * 32 + lane];
    unsigned m = __builtin_amdgcn_ballot_w32(v > 0);
    if (lane == 0) bits[(size_t)row * 128 + c] = m;
}

// ---- Kernel 1: f32 -> f16 row-major convert ----
__global__ void gat_f32_to_f16(const float* __restrict__ src, _Float16* __restrict__ dst, int n) {
    int i = blockIdx.x * blockDim.x + threadIdx.x;
    if (i < n) dst[i] = (_Float16)src[i];
}

// ---- Kernel 2: pack f32 row-major matrix (K x Ncols at colOff, leading dim ld)
// into WMMA B-operand swizzled f16 ----
__global__ void gat_pack_b(const float* __restrict__ src, int ld, int colOff,
                           _Float16* __restrict__ dst, int ntTotal, int ntBase,
                           int ntLocal, int ktCount) {
    int t = blockIdx.x * blockDim.x + threadIdx.x;
    int total = ktCount * ntLocal * 32;
    if (t >= total) return;
    int lane = t & 31;
    int tmp  = t >> 5;
    int ntL  = tmp % ntLocal;
    int kt   = tmp / ntLocal;
    int hi   = lane >> 4;
    int col  = colOff + ntL * 16 + (lane & 15);
    int kb   = kt * 32 + hi * 8;
    _Float16* d = dst + (((size_t)kt * ntTotal + ntBase + ntL) * 32 + lane) * 16;
    #pragma unroll
    for (int i = 0; i < 16; ++i)
        d[i] = (_Float16)src[(size_t)(kb + ((i < 8) ? i : (i + 8))) * ld + col];
}

// ---- Kernel 3: WMMA GEMM  C(f32, M x 16*ntTotal) = A(f16 row-major) @ Bpacked ----
__global__ __launch_bounds__(32) void gat_gemm_wmma(
    const _Float16* __restrict__ A, int lda,
    const _Float16* __restrict__ Bp, int ntTotal,
    float* __restrict__ C, int ldc, int ktCount)
{
    int lane = threadIdx.x;
    int hi   = lane >> 4;
    int r    = lane & 15;
    int ntB  = blockIdx.x;
    int mt   = blockIdx.y;
    v8f acc = {};
    const _Float16* arow = A + (size_t)(mt * 16 + r) * lda;
    for (int kt = 0; kt < ktCount; ++kt) {
        int kb = kt * 32 + hi * 8;
        v16h a = load16h2(arow + kb, arow + kb + 16);
        const _Float16* bp = Bp + (((size_t)kt * ntTotal + ntB) * 32 + lane) * 16;
        __builtin_prefetch(bp + (size_t)ntTotal * 512, 0, 3);   // next kt's B tile
        v16h b = load16h(bp);
        acc = WMMA_F16(a, b, acc);
    }
    float* crow = C + (size_t)(mt * 16 + hi * 8) * ldc + ntB * 16 + r;
    #pragma unroll
    for (int v = 0; v < 8; ++v)
        crow[(size_t)v * ldc] = acc[v];
}

// ---- Kernel 4: attention projections  aL[n]=h[n]·a[:F], aR[n]=h[n]·a[F:] ----
__global__ void gat_alr(const float* __restrict__ H, int ld, int colOff,
                        const float* __restrict__ avec,
                        float* __restrict__ aL, float* __restrict__ aR, int n, int F) {
    int i = blockIdx.x * blockDim.x + threadIdx.x;
    if (i >= n) return;
    const float* row = H + (size_t)i * ld + colOff;
    float sl = 0.f, sr = 0.f;
    for (int f = 0; f < F; ++f) {
        float h = row[f];
        sl += h * avec[f];
        sr += h * avec[F + f];
    }
    aL[i] = sl;
    aR[i] = sr;
}

// ---- Kernel 5: exact per-row softmax max.
// leakyrelu is monotonic => m_i = leakyrelu(aL[i] + max_{j in adj(i)} aR[j]).
// One wave per row; lane scans 4 of the 128 bitmask words. ----
__global__ __launch_bounds__(32) void gat_rowmax(
    const float* __restrict__ aL, const float* __restrict__ aR,
    const unsigned* __restrict__ adjbits, float* __restrict__ rowM) {
    int row  = blockIdx.x;
    int lane = threadIdx.x;
    const unsigned* mrow = adjbits + (size_t)row * 128;
    float mx = -3.0e38f;
    #pragma unroll
    for (int t = 0; t < 4; ++t) {
        int jt = lane + t * 32;
        unsigned mask = mrow[jt];
        for (int b = 0; b < 32; ++b) {
            float v = ((mask >> b) & 1u) ? aR[jt * 32 + b] : -3.0e38f;
            mx = fmaxf(mx, v);
        }
    }
    #pragma unroll
    for (int off = 16; off >= 1; off >>= 1)
        mx = fmaxf(mx, __shfl_xor(mx, off, 32));
    float e = aL[row] + mx;
    e = (e >= 0.f) ? e : GAT_ALPHA * e;
    if (lane == 0) rowM[row] = e;
}

// ---- Kernel 6: fused masked-softmax attention: out = softmax(mask(e)) @ h
// One wave per 32-row block (2 A fragments, 8 accumulators); exact precomputed
// row max => no online rescale; 4 B-frag loads feed 8 WMMAs per 32-j step. ----
__global__ __launch_bounds__(32) void gat_attn(
    const float* __restrict__ aL, const float* __restrict__ aR,
    const float* __restrict__ rowM,
    const unsigned* __restrict__ adjbits,
    const _Float16* __restrict__ Bp,            // packed h: ntTotal=4, K=4096
    float* __restrict__ outF, int outLdF, int outOffF,
    _Float16* __restrict__ outH, int outLdH, int outOffH,
    int applyElu)
{
    int lane = threadIdx.x;
    int hi   = lane >> 4;
    int r    = lane & 15;
    int rowBase = blockIdx.x * 32;
    float aL0 = aL[rowBase + r];
    float aL1 = aL[rowBase + 16 + r];
    float m0  = rowM[rowBase + r];
    float m1  = rowM[rowBase + 16 + r];
    const unsigned* mr0 = adjbits + (size_t)(rowBase + r) * 128;
    const unsigned* mr1 = mr0 + 16 * 128;

    v8f a00 = {}, a01 = {}, a02 = {}, a03 = {};
    v8f a10 = {}, a11 = {}, a12 = {}, a13 = {};
    float ps0 = 0.f, ps1 = 0.f;

    for (int jt = 0; jt < GAT_N / 32; ++jt) {
        unsigned mk0 = mr0[jt];
        unsigned mk1 = mr1[jt];
        v16h f0, f1;
        #pragma unroll
        for (int i = 0; i < 16; ++i) {
            int kk = a_kidx(i, hi);                 // j within the 32-block
            float ar = aR[jt * 32 + kk];
            float e0 = aL0 + ar; e0 = (e0 >= 0.f) ? e0 : GAT_ALPHA * e0;
            float e1 = aL1 + ar; e1 = (e1 >= 0.f) ? e1 : GAT_ALPHA * e1;
            float p0 = ((mk0 >> kk) & 1u) ? __expf(e0 - m0) : 0.f;
            float p1 = ((mk1 >> kk) & 1u) ? __expf(e1 - m1) : 0.f;
            ps0 += p0;  ps1 += p1;
            f0[i] = (_Float16)p0;
            f1[i] = (_Float16)p1;
        }
        const _Float16* bb = Bp + ((size_t)jt * 128 + lane) * 16;
        __builtin_prefetch(bb + 128 * 16, 0, 3);    // next j-block's B tiles
        v16h b0 = load16h(bb);
        v16h b1 = load16h(bb + 512);
        v16h b2 = load16h(bb + 1024);
        v16h b3 = load16h(bb + 1536);
        a00 = WMMA_F16(f0, b0, a00);
        a01 = WMMA_F16(f0, b1, a01);
        a02 = WMMA_F16(f0, b2, a02);
        a03 = WMMA_F16(f0, b3, a03);
        a10 = WMMA_F16(f1, b0, a10);
        a11 = WMMA_F16(f1, b1, a11);
        a12 = WMMA_F16(f1, b2, a12);
        a13 = WMMA_F16(f1, b3, a13);
    }

    ps0 += __shfl_xor(ps0, 16, 32);                 // full row sums
    ps1 += __shfl_xor(ps1, 16, 32);
    float li0 = 1.0f / ps0;
    float li1 = 1.0f / ps1;

    #pragma unroll
    for (int blk = 0; blk < 2; ++blk) {
        float li = blk ? li1 : li0;
        int base = rowBase + blk * 16;
        #pragma unroll
        for (int v = 0; v < 8; ++v) {
            float sv = __shfl(li, v + hi * 8, 32);  // row (v+8*hi)'s 1/l
            float o0 = (blk ? a10[v] : a00[v]) * sv;
            float o1 = (blk ? a11[v] : a01[v]) * sv;
            float o2 = (blk ? a12[v] : a02[v]) * sv;
            float o3 = (blk ? a13[v] : a03[v]) * sv;
            if (applyElu) {
                o0 = (o0 > 0.f) ? o0 : __expf(o0) - 1.f;
                o1 = (o1 > 0.f) ? o1 : __expf(o1) - 1.f;
                o2 = (o2 > 0.f) ? o2 : __expf(o2) - 1.f;
                o3 = (o3 > 0.f) ? o3 : __expf(o3) - 1.f;
            }
            int row = base + v + hi * 8;
            if (outF) {
                float* o = outF + (size_t)row * outLdF + outOffF + r;
                o[0] = o0; o[16] = o1; o[32] = o2; o[48] = o3;
            }
            if (outH) {
                _Float16* o = outH + (size_t)row * outLdH + outOffH + r;
                o[0] = (_Float16)o0; o[16] = (_Float16)o1;
                o[32] = (_Float16)o2; o[48] = (_Float16)o3;
            }
        }
    }
}

extern "C" void kernel_launch(void* const* d_in, const int* in_sizes, int n_in,
                              void* d_out, int out_size, void* d_ws, size_t ws_size,
                              hipStream_t stream) {
    const float* features = (const float*)d_in[0];   // 4096 x 256
    const int*   adj      = (const int*)  d_in[1];   // 4096 x 4096
    const float* W_h      = (const float*)d_in[2];   // 8 x 256 x 64
    const float* a_h      = (const float*)d_in[3];   // 8 x 128 x 1
    const float* W_out    = (const float*)d_in[4];   // 512 x 64
    const float* a_out    = (const float*)d_in[5];   // 128 x 1
    float* out = (float*)d_out;                      // 4096 x 64

    // ---- carve workspace (all offsets 256B aligned) ----
    char* ws = (char*)d_ws;
    auto carve = [&](size_t bytes) { char* p = ws; ws += (bytes + 255) & ~(size_t)255; return p; };
    _Float16* featH    = (_Float16*)carve((size_t)4096 * 256 * 2);
    _Float16* WcatP    = (_Float16*)carve((size_t)8 * 32 * 32 * 16 * 2);
    _Float16* WoutP    = (_Float16*)carve((size_t)16 * 4 * 32 * 16 * 2);
    unsigned* adjb     = (unsigned*)carve((size_t)4096 * 128 * 4);
    float*    Hmat     = (float*)   carve((size_t)4096 * 512 * 4);
    _Float16* packedH  = (_Float16*)carve((size_t)8 * 128 * 4 * 32 * 16 * 2);
    float*    aL1      = (float*)   carve((size_t)8 * 4096 * 4);
    float*    aR1      = (float*)   carve((size_t)8 * 4096 * 4);
    float*    rowM1    = (float*)   carve((size_t)8 * 4096 * 4);
    _Float16* XcatH    = (_Float16*)carve((size_t)4096 * 512 * 2);
    float*    H2       = (float*)   carve((size_t)4096 * 64 * 4);
    _Float16* packedH2 = (_Float16*)carve((size_t)128 * 4 * 32 * 16 * 2);
    float*    aL2      = (float*)   carve((size_t)4096 * 4);
    float*    aR2      = (float*)   carve((size_t)4096 * 4);
    float*    rowM2    = (float*)   carve((size_t)4096 * 4);
    (void)ws_size; (void)in_sizes; (void)n_in; (void)out_size;

    // 0) adjacency -> bitmask (64 MB read once; attention passes read 2 MB from L2)
    gat_pack_adj<<<(4096 * 128 * 32) / 256, 256, 0, stream>>>(adj, adjb);
    // 1) features -> f16
    gat_f32_to_f16<<<(4096 * 256 + 255) / 256, 256, 0, stream>>>(features, featH, 4096 * 256);
    // 2) pack weights into WMMA B-operand layout
    for (int h = 0; h < 8; ++h)
        gat_pack_b<<<(8 * 4 * 32 + 255) / 256, 256, 0, stream>>>(
            W_h + (size_t)h * 256 * 64, 64, 0, WcatP, 32, h * 4, 4, 8);
    gat_pack_b<<<(16 * 4 * 32 + 255) / 256, 256, 0, stream>>>(W_out, 64, 0, WoutP, 4, 0, 4, 16);
    // 3) Hmat = X @ [W_0 | ... | W_7]   (4096x256 @ 256x512)
    gat_gemm_wmma<<<dim3(32, 256), 32, 0, stream>>>(featH, 256, WcatP, 32, Hmat, 512, 8);
    // 4) per-head projections, exact row maxes, pack h as B operand
    for (int h = 0; h < 8; ++h) {
        gat_alr<<<16, 256, 0, stream>>>(Hmat, 512, h * 64, a_h + (size_t)h * 128,
                                        aL1 + (size_t)h * 4096, aR1 + (size_t)h * 4096, 4096, 64);
        gat_rowmax<<<4096, 32, 0, stream>>>(aL1 + (size_t)h * 4096, aR1 + (size_t)h * 4096,
                                            adjb, rowM1 + (size_t)h * 4096);
        gat_pack_b<<<(128 * 4 * 32 + 255) / 256, 256, 0, stream>>>(
            Hmat, 512, h * 64, packedH + (size_t)h * 128 * 4 * 32 * 16, 4, 0, 4, 128);
    }
    // 5) per-head fused masked-softmax attention -> elu -> XcatH (f16)
    for (int h = 0; h < 8; ++h)
        gat_attn<<<128, 32, 0, stream>>>(aL1 + (size_t)h * 4096, aR1 + (size_t)h * 4096,
                                         rowM1 + (size_t)h * 4096, adjb,
                                         packedH + (size_t)h * 128 * 4 * 32 * 16,
                                         nullptr, 0, 0, XcatH, 512, h * 64, 1);
    // 6) H2 = Xcat @ W_out   (4096x512 @ 512x64)
    gat_gemm_wmma<<<dim3(4, 256), 32, 0, stream>>>(XcatH, 512, WoutP, 4, H2, 64, 16);
    // 7) output-layer projections + row max + pack
    gat_alr<<<16, 256, 0, stream>>>(H2, 64, 0, a_out, aL2, aR2, 4096, 64);
    gat_rowmax<<<4096, 32, 0, stream>>>(aL2, aR2, adjb, rowM2);
    gat_pack_b<<<(128 * 4 * 32 + 255) / 256, 256, 0, stream>>>(H2, 64, 0, packedH2, 4, 0, 4, 128);
    // 8) final attention (no elu) -> d_out (f32, 4096x64)
    gat_attn<<<128, 32, 0, stream>>>(aL2, aR2, rowM2, adjb, packedH2, out, 64, 0, nullptr, 0, 0, 0);
}
